// TransformerFuser_75600014344418
// MI455X (gfx1250) — compile-verified, matrix operand order checked by reading, not run on hardware
//
#include <hip/hip_runtime.h>
#include <hip/hip_bf16.h>

// ---------------------------------------------------------------------------
// TransformerFuser for MI455X (gfx1250, wave32).
// GEMMs + attention matmuls on v_wmma_f32_16x16x32_bf16, bf16 staged in LDS
// (conversion off the critical path), QKV stored bf16 in global, async
// global->LDS staging (ASYNCcnt) via __builtin_amdgcn_global_load_async_to_lds_b128.
// ---------------------------------------------------------------------------

#define BSZ 64
#define SEQ 256
#define MD  768
#define NH  12
#define DKH 64
#define FFD 3072
#define ROWS (BSZ * SEQ)   // 16384

typedef __attribute__((ext_vector_type(16))) __bf16 v16bf;
typedef __attribute__((ext_vector_type(8)))  __bf16 v8bf;
typedef __attribute__((ext_vector_type(2)))  __bf16 v2bf;
typedef __attribute__((ext_vector_type(8)))  float  v8f;
typedef int v4i_vs __attribute__((vector_size(16)));   // matches builtin param

// ---- async global->LDS staging (gfx1250 ASYNCcnt path), guarded ----------
#if defined(__has_builtin)
#if __has_builtin(__builtin_amdgcn_global_load_async_to_lds_b128)
#define ASYNC_LDS 1
#endif
#endif

#if defined(ASYNC_LDS)
#define ASYNC_COPY16(gp, lp)                                                   \
  __builtin_amdgcn_global_load_async_to_lds_b128(                              \
      (__attribute__((address_space(1))) v4i_vs*)(unsigned long long)(const void*)(gp), \
      (__attribute__((address_space(3))) v4i_vs*)(unsigned int)(unsigned long long)(const void*)(lp), \
      0, 0)
#if __has_builtin(__builtin_amdgcn_s_wait_asynccnt)
#define ASYNC_WAIT() __builtin_amdgcn_s_wait_asynccnt(0)
#else
#define ASYNC_WAIT() asm volatile("s_wait_asynccnt 0x0" ::: "memory")
#endif
#else
#define ASYNC_COPY16(gp, lp) (*(v8bf*)(lp) = *(const v8bf*)(gp))
#define ASYNC_WAIT() ((void)0)
#endif

__device__ __forceinline__ v8f wmma_bf16(v16bf a, v16bf b, v8f c) {
  return __builtin_amdgcn_wmma_f32_16x16x32_bf16(
      false, a, false, b, (short)0, c, false, false);
}

// Build a 16x32 A-style fragment (or 32x16 B from a transposed row):
// lane half selects K sub-blocks {khalf..khalf+7, khalf+16..khalf+23}.
// Two 16-byte LDS loads, no conversions.
__device__ __forceinline__ v16bf frag_from(const __bf16* rowbase, int khalf) {
  v8bf lo = *(const v8bf*)(rowbase + khalf);
  v8bf hi = *(const v8bf*)(rowbase + khalf + 16);
  return __builtin_shufflevector(lo, hi, 0, 1, 2, 3, 4, 5, 6, 7,
                                 8, 9, 10, 11, 12, 13, 14, 15);
}

// ---------------------------------------------------------------------------
// Generic GEMM: C[M,N] = act(A[M,K] @ W[K,N] + bias[N] (+ R[M,N]))
// 256 threads (8 waves), block tile 64(M) x 256(N), wave tile 32x64,
// 8 WMMAs per 32-wide K step. bf16 staged in LDS (B transposed), fp32 accum.
// Output fp32 or bf16 (obf). N % 256 == 0 (768/3072); M, K arbitrary.
// ---------------------------------------------------------------------------
__global__ __launch_bounds__(256)
void gemm_bias_act(const float* __restrict__ A, const float* __restrict__ W,
                   const float* __restrict__ bias, const float* __restrict__ R,
                   void* __restrict__ Cout, int M, int N, int K, int relu, int obf)
{
  __shared__ __attribute__((aligned(16))) __bf16 As[64][40];   // pad: 80B rows
  __shared__ __attribute__((aligned(16))) __bf16 Bt[256][40];  // [n][k]
  const int tid  = threadIdx.x;
  const int lane = tid & 31;
  const int wave = tid >> 5;
  const int wm   = wave >> 2;          // 2 M-wave-tiles of 32
  const int wn   = wave & 3;           // 4 N-wave-tiles of 64
  const int m0   = blockIdx.y * 64;
  const int n0   = blockIdx.x * 256;
  const int row   = lane & 15;
  const int khalf = (lane >> 4) << 3;

  v8f acc[2][4] = {};

  for (int k0 = 0; k0 < K; k0 += 32) {
    // Stage A 64x32: fp32 -> packed bf16 pairs (v_cvt_pk_bf16_f32 + b32 store)
    for (int i = tid; i < 64 * 16; i += 256) {
      int r = i >> 4, c = (i & 15) << 1;
      int gm = m0 + r, gk = k0 + c;
      float x0 = (gm < M && gk < K)     ? A[(size_t)gm * K + gk]     : 0.0f;
      float x1 = (gm < M && gk + 1 < K) ? A[(size_t)gm * K + gk + 1] : 0.0f;
      v2bf p = {(__bf16)x0, (__bf16)x1};
      *(v2bf*)&As[r][c] = p;
    }
    // Stage B 32x256 transposed into Bt[n][k] (pairs along K)
    for (int i = tid; i < 16 * 256; i += 256) {
      int rr = (i >> 8) << 1, c = i & 255;
      int gk = k0 + rr, gn = n0 + c;
      float w0 = (gk < K && gn < N)     ? W[(size_t)gk * N + gn]       : 0.0f;
      float w1 = (gk + 1 < K && gn < N) ? W[(size_t)(gk + 1) * N + gn] : 0.0f;
      v2bf p = {(__bf16)w0, (__bf16)w1};
      *(v2bf*)&Bt[c][rr] = p;
    }
    // Prefetch next K tile (global_prefetch_b8)
    if (k0 + 32 < K) {
      int pk = k0 + 32;
      int pr = tid >> 2;
      if (m0 + pr < M) __builtin_prefetch(&A[(size_t)(m0 + pr) * K + pk], 0, 1);
      int br = tid >> 3;
      if (pk + br < K) __builtin_prefetch(&W[(size_t)(pk + br) * N + n0 + (tid & 7) * 32], 0, 1);
    }
    __syncthreads();

    v16bf af0 = frag_from(&As[wm * 32 + row][0], khalf);
    v16bf af1 = frag_from(&As[wm * 32 + 16 + row][0], khalf);
    #pragma unroll
    for (int t = 0; t < 4; ++t) {
      v16bf bf = frag_from(&Bt[wn * 64 + t * 16 + row][0], khalf);
      acc[0][t] = wmma_bf16(af0, bf, acc[0][t]);
      acc[1][t] = wmma_bf16(af1, bf, acc[1][t]);
    }
    __syncthreads();
  }

  // Epilogue. C layout: VGPR r -> M = r + 8*(lane>=16), N = lane&15.
  const int rb0 = m0 + wm * 32 + ((lane >> 4) << 3);
  #pragma unroll
  for (int tm = 0; tm < 2; ++tm) {
    #pragma unroll
    for (int t = 0; t < 4; ++t) {
      const int cn = n0 + wn * 64 + t * 16 + row;
      if (cn < N) {
        const float bv = bias[cn];
        #pragma unroll
        for (int r = 0; r < 8; ++r) {
          const int gm = rb0 + tm * 16 + r;
          if (gm < M) {
            float v = acc[tm][t][r] + bv;
            if (R)    v += R[(size_t)gm * N + cn];
            if (relu) v = fmaxf(v, 0.0f);
            if (obf)  ((__bf16*)Cout)[(size_t)gm * N + cn] = (__bf16)v;
            else      ((float*)Cout)[(size_t)gm * N + cn]  = v;
          }
        }
      }
    }
  }
}

// ---------------------------------------------------------------------------
// LayerNorm over last dim (768). One wave per row; in-place safe.
// ---------------------------------------------------------------------------
__global__ __launch_bounds__(256)
void layernorm_rows(const float* __restrict__ X, const float* __restrict__ g,
                    const float* __restrict__ b, float* __restrict__ Y, int rows)
{
  const int lane = threadIdx.x & 31;
  const int wave = threadIdx.x >> 5;
  const int row  = blockIdx.x * 8 + wave;
  if (row >= rows) return;
  const float* x = X + (size_t)row * MD;
  float v[24];
  float s = 0.0f;
  #pragma unroll
  for (int i = 0; i < 24; ++i) { v[i] = x[i * 32 + lane]; s += v[i]; }
  #pragma unroll
  for (int o = 16; o > 0; o >>= 1) s += __shfl_xor(s, o, 32);
  const float mean = s * (1.0f / 768.0f);
  float q = 0.0f;
  #pragma unroll
  for (int i = 0; i < 24; ++i) { float d = v[i] - mean; q += d * d; }
  #pragma unroll
  for (int o = 16; o > 0; o >>= 1) q += __shfl_xor(q, o, 32);
  const float rstd = rsqrtf(q * (1.0f / 768.0f) + 1e-5f);
  float* y = Y + (size_t)row * MD;
  #pragma unroll
  for (int i = 0; i < 24; ++i) {
    int c = i * 32 + lane;
    y[c] = (v[i] - mean) * rstd * g[c] + b[c];
  }
}

__global__ void nonempty_k(const int* __restrict__ masks, int* __restrict__ ne)
{
  __shared__ int s;
  if (threadIdx.x == 0) s = 0;
  __syncthreads();
  if (masks[(size_t)blockIdx.x * SEQ + threadIdx.x] != 0) atomicOr(&s, 1);
  __syncthreads();
  if (threadIdx.x == 0) ne[blockIdx.x] = s;
}

// ---------------------------------------------------------------------------
// Self-attention: block (128 thr, 4 waves) per (b, h, 16-query tile).
// Q/K/V arrive as bf16 (projection GEMMs emit bf16): K and Q staged to LDS
// via async b128 copies (ASYNCcnt); V staged transposed so P@V B-fragments
// are contiguous. Scores & P@V on WMMA.
// ---------------------------------------------------------------------------
__global__ __launch_bounds__(128)
void self_attn_k(const __bf16* __restrict__ Q, const __bf16* __restrict__ K,
                 const __bf16* __restrict__ V, const int* __restrict__ ne,
                 float* __restrict__ O)
{
  const int qt = blockIdx.x, h = blockIdx.y, b = blockIdx.z;
  const int tid = threadIdx.x, lane = tid & 31, wave = tid >> 5;
  __shared__ __attribute__((aligned(16))) __bf16 Kb[SEQ][72];   // 36.9 KB
  __shared__ __attribute__((aligned(16))) __bf16 Vt[DKH][264];  // 33.8 KB
  __shared__ __attribute__((aligned(16))) __bf16 Qs[16][72];
  __shared__ __attribute__((aligned(16))) float  Ss[16][SEQ];   // 16.4 KB
  __shared__ __attribute__((aligned(16))) __bf16 Pb[16][264];

  const size_t hbase = (size_t)(b * SEQ) * MD + (size_t)h * DKH;

  // K rows -> LDS (8 x 16B chunks per row), async ASYNCcnt path
  for (int i = tid; i < SEQ * 8; i += 128) {
    int j = i >> 3, c = (i & 7) << 3;
    ASYNC_COPY16(K + hbase + (size_t)j * MD + c, &Kb[j][c]);
  }
  // Q tile (16 rows x 8 chunks = 128 chunks, one per thread)
  {
    int j = tid >> 3, c = (tid & 7) << 3;
    ASYNC_COPY16(Q + hbase + (size_t)(qt * 16 + j) * MD + c, &Qs[j][c]);
  }
  // V staged transposed: Vt[d][j] (coalesced b128 global reads)
  for (int i = tid; i < SEQ * 8; i += 128) {
    int j = i >> 3, c = (i & 7) << 3;
    v8bf v = *(const v8bf*)(V + hbase + (size_t)j * MD + c);
    #pragma unroll
    for (int k = 0; k < 8; ++k) Vt[c + k][j] = v[k];
  }
  ASYNC_WAIT();
  __syncthreads();

  const int row   = lane & 15;
  const int khalf = (lane >> 4) << 3;

  // ---- scores: S(16x256) = Q(16x64) @ K^T, 2 WMMAs per 16-key block ----
  v16bf a0 = frag_from(&Qs[row][0],  khalf);
  v16bf a1 = frag_from(&Qs[row][32], khalf);
  for (int jb = wave; jb < 16; jb += 4) {
    const int n = jb * 16 + row;
    v16bf b0 = frag_from(&Kb[n][0],  khalf);
    v16bf b1 = frag_from(&Kb[n][32], khalf);
    v8f acc = {};
    acc = wmma_bf16(a0, b0, acc);
    acc = wmma_bf16(a1, b1, acc);
    const int rb = (lane >> 4) << 3;
    #pragma unroll
    for (int r = 0; r < 8; ++r) Ss[rb + r][jb * 16 + row] = acc[r] * 0.125f;
  }
  __syncthreads();

  // ---- masked stable softmax (wave handles rows wave, wave+4, ...) ----
  const bool valid = (ne[b] != 0);
  for (int r = wave; r < 16; r += 4) {
    const int qrow = qt * 16 + r;
    float vb[8];
    float m = -3.0e38f;
    #pragma unroll
    for (int t = 0; t < 8; ++t) {
      int j = t * 32 + lane;
      float s = Ss[r][j];
      if (!(valid && j <= qrow)) s = -1.0e9f;   // all-masked row -> uniform
      vb[t] = s;
      m = fmaxf(m, s);
    }
    #pragma unroll
    for (int o = 16; o > 0; o >>= 1) m = fmaxf(m, __shfl_xor(m, o, 32));
    float sum = 0.0f;
    #pragma unroll
    for (int t = 0; t < 8; ++t) { vb[t] = __expf(vb[t] - m); sum += vb[t]; }
    #pragma unroll
    for (int o = 16; o > 0; o >>= 1) sum += __shfl_xor(sum, o, 32);
    const float inv = 1.0f / sum;
    #pragma unroll
    for (int t = 0; t < 8; ++t) Pb[r][t * 32 + lane] = (__bf16)(vb[t] * inv);
  }
  __syncthreads();

  // ---- out(16x64) = P(16x256) @ V(256x64): one 16-col block per wave ----
  const int nb = wave;
  v8f acc = {};
  #pragma unroll
  for (int kc = 0; kc < 8; ++kc) {
    v16bf a  = frag_from(&Pb[row][kc * 32], khalf);
    v16bf bf = frag_from(&Vt[nb * 16 + row][kc * 32], khalf);
    acc = wmma_bf16(a, bf, acc);
  }
  const int col = h * DKH + nb * 16 + row;
  const int rb  = (lane >> 4) << 3;
  #pragma unroll
  for (int r = 0; r < 8; ++r) {
    int qrow = qt * 16 + rb + r;
    O[(size_t)(b * SEQ + qrow) * MD + col] = acc[r];
  }
}

// ---------------------------------------------------------------------------
// Cross-attention: one broadcast query per (b,h); causal softmax over the
// same score vector => running prefix scan. One wave per (b,h).
// ---------------------------------------------------------------------------
__global__ __launch_bounds__(256)
void cross_attn_k(const __bf16* __restrict__ QH, const __bf16* __restrict__ K,
                  const __bf16* __restrict__ V, const int* __restrict__ ne,
                  float* __restrict__ O)
{
  __shared__ float qs[8][DKH];
  const int lane = threadIdx.x & 31;
  const int wv   = threadIdx.x >> 5;
  const int wid  = blockIdx.x * 8 + wv;          // (b,h), grid = 96
  const int b = wid / NH, h = wid - b * NH;
  const size_t hbase = (size_t)(b * SEQ) * MD + (size_t)h * DKH;

  qs[wv][lane]      = (float)QH[(size_t)b * MD + h * DKH + lane];
  qs[wv][lane + 32] = (float)QH[(size_t)b * MD + h * DKH + lane + 32];

  float sv[8];
  #pragma unroll
  for (int t = 0; t < 8; ++t) {
    const int j = t * 32 + lane;
    const __bf16* kr = K + hbase + (size_t)j * MD;
    float s = 0.0f;
    #pragma unroll 8
    for (int d = 0; d < DKH; ++d) s += qs[wv][d] * (float)kr[d];
    sv[t] = s * 0.125f;
  }
  float m = sv[0];
  #pragma unroll
  for (int t = 1; t < 8; ++t) m = fmaxf(m, sv[t]);
  #pragma unroll
  for (int o = 16; o > 0; o >>= 1) m = fmaxf(m, __shfl_xor(m, o, 32));

  float n0 = 0.0f, n1 = 0.0f, den = 0.0f;
  if (ne[b] != 0) {
    for (int j = 0; j < SEQ; ++j) {
      const float sj = __shfl(sv[j >> 5], j & 31, 32);
      const float e  = __expf(sj - m);
      const __bf16* vr = V + hbase + (size_t)j * MD;
      n0 += e * (float)vr[lane];
      n1 += e * (float)vr[lane + 32];
      den += e;
      const float inv = 1.0f / den;
      float* orow = O + hbase + (size_t)j * MD;
      orow[lane]      = n0 * inv;
      orow[lane + 32] = n1 * inv;
    }
  } else {
    for (int j = 0; j < SEQ; ++j) {
      const __bf16* vr = V + hbase + (size_t)j * MD;
      n0 += (float)vr[lane];
      n1 += (float)vr[lane + 32];
    }
    const float inv = 1.0f / (float)SEQ;
    for (int j = 0; j < SEQ; ++j) {
      float* orow = O + hbase + (size_t)j * MD;
      orow[lane]      = n0 * inv;
      orow[lane + 32] = n1 * inv;
    }
  }
}

// ---------------------------------------------------------------------------
// Host orchestration
// ---------------------------------------------------------------------------
static inline void launch_gemm(const float* A, const float* W, const float* bias,
                               const float* R, void* C, int M, int N, int K,
                               int relu, int obf, hipStream_t s)
{
  dim3 grid(N / 256, (M + 63) / 64);
  gemm_bias_act<<<grid, 256, 0, s>>>(A, W, bias, R, C, M, N, K, relu, obf);
}

static inline void launch_ln(const float* X, const float* g, const float* b,
                             float* Y, int rows, hipStream_t s)
{
  layernorm_rows<<<(rows + 7) / 8, 256, 0, s>>>(X, g, b, Y, rows);
}

extern "C" void kernel_launch(void* const* d_in, const int* in_sizes, int n_in,
                              void* d_out, int out_size, void* d_ws, size_t ws_size,
                              hipStream_t stream)
{
  (void)in_sizes; (void)n_in; (void)out_size; (void)ws_size;
  const float* inputs = (const float*)d_in[0];   // (B, D, 782)
  const int*   rmask  = (const int*)d_in[1];     // (B, D)
  const float* raud   = (const float*)d_in[2];   // (B, 768)
  auto P = [&](int i) { return (const float*)d_in[i]; };
  const float* audio_W = P(3);
  const float* audio_b = P(4);
  const float* feat_W  = P(5);
  const float* feat_b  = P(6);
  const float* fin_b   = P(7);
  const float* fin_g   = P(8);

  const size_t SZ = (size_t)ROWS * MD;
  float* X   = (float*)d_ws;
  float* NX  = X  + SZ;
  float* Qb  = NX + SZ;                          // bf16 payload (half used)
  float* Kb  = Qb + SZ;
  float* Vb  = Kb + SZ;
  float* AO  = Vb + SZ;
  float* HB  = AO + SZ;                          // (ROWS, 3072) fp32
  float* QV  = HB + (size_t)ROWS * FFD;          // (B, 768) fp32 audio proj
  float* QH  = QV + (size_t)BSZ * MD;            // (B, 768) bf16 ca query
  int*   NE  = (int*)(QH + (size_t)BSZ * MD);

  nonempty_k<<<BSZ, SEQ, 0, stream>>>(rmask, NE);
  launch_gemm(inputs, feat_W, feat_b, nullptr, X, ROWS, MD, MD + 14, 0, 0, stream);
  launch_gemm(raud, audio_W, audio_b, nullptr, QV, BSZ, MD, MD, 0, 0, stream);

  for (int l = 0; l < 2; ++l) {
    const int base = 9 + 26 * l;
    const float *ca_Wk = P(base + 0),  *ca_Wo = P(base + 1);
    const float *ca_Wq = P(base + 2),  *ca_Wv = P(base + 3);
    const float *ca_bk = P(base + 4),  *ca_bo = P(base + 5);
    const float *ca_bq = P(base + 6),  *ca_bv = P(base + 7);
    const float *fW1   = P(base + 8),  *fW2   = P(base + 9);
    const float *fb1   = P(base + 10), *fb2   = P(base + 11);
    const float *ln0_b = P(base + 12), *ln0_g = P(base + 13);
    const float *ln1_b = P(base + 14), *ln1_g = P(base + 15);
    const float *lnm_b = P(base + 16), *lnm_g = P(base + 17);
    const float *sa_Wk = P(base + 18), *sa_Wo = P(base + 19);
    const float *sa_Wq = P(base + 20), *sa_Wv = P(base + 21);
    const float *sa_bk = P(base + 22), *sa_bo = P(base + 23);
    const float *sa_bq = P(base + 24), *sa_bv = P(base + 25);

    // ---- self-attention sublayer (pre-norm residual) ----
    launch_ln(X, ln0_g, ln0_b, NX, ROWS, stream);
    launch_gemm(NX, sa_Wq, sa_bq, nullptr, Qb, ROWS, MD, MD, 0, 1, stream);
    launch_gemm(NX, sa_Wk, sa_bk, nullptr, Kb, ROWS, MD, MD, 0, 1, stream);
    launch_gemm(NX, sa_Wv, sa_bv, nullptr, Vb, ROWS, MD, MD, 0, 1, stream);
    self_attn_k<<<dim3(SEQ / 16, NH, BSZ), 128, 0, stream>>>(
        (const __bf16*)Qb, (const __bf16*)Kb, (const __bf16*)Vb, NE, AO);
    launch_gemm(AO, sa_Wo, sa_bo, X, X, ROWS, MD, MD, 0, 0, stream);

    // ---- cross-attention (single audio query, causal broadcast) ----
    launch_gemm(QV, ca_Wq, ca_bq, nullptr, QH, BSZ, MD, MD, 0, 1, stream);
    launch_gemm(X, ca_Wk, ca_bk, nullptr, Kb, ROWS, MD, MD, 0, 1, stream);
    launch_gemm(X, ca_Wv, ca_bv, nullptr, Vb, ROWS, MD, MD, 0, 1, stream);
    cross_attn_k<<<(BSZ * NH) / 8, 256, 0, stream>>>(
        (const __bf16*)QH, (const __bf16*)Kb, (const __bf16*)Vb, NE, AO);
    launch_gemm(AO, ca_Wo, ca_bo, X, X, ROWS, MD, MD, 0, 0, stream);

    // ---- mid LN + FFN ----
    launch_ln(X, lnm_g, lnm_b, X, ROWS, stream);
    launch_ln(X, ln1_g, ln1_b, NX, ROWS, stream);
    launch_gemm(NX, fW1, fb1, nullptr, HB, ROWS, FFD, MD, 1, 0, stream);
    launch_gemm(HB, fW2, fb2, X, X, ROWS, MD, FFD, 0, 0, stream);
  }
  launch_ln(X, fin_g, fin_b, (float*)d_out, ROWS, stream);
}
